// NaiveFourierKANLayer_80333068304841
// MI455X (gfx1250) — matrix operand order repaired and checked by reading, not compiled, not run on hardware
//
#include <hip/hip_runtime.h>

typedef __attribute__((ext_vector_type(16))) _Float16 v16h;
typedef __attribute__((ext_vector_type(8)))  float    v8f;

#define IN_DIM   64
#define OUT_DIM  64
#define KGRID    100
#define KPAD     112   // 7 * 16, padded harmonic count
#define STEPS    7     // KPAD / 16 k-values per WMMA step (x2 for cos+sin = K=32)

// ---- prep: fp32 coeffs [2][J][I][100] -> f16 [2][J][I][112], zero padded ----
__global__ void fkan_prep(const float* __restrict__ fc, _Float16* __restrict__ w, int total) {
    int idx = blockIdx.x * blockDim.x + threadIdx.x;
    if (idx >= total) return;
    int k    = idx % KPAD;
    int rest = idx / KPAD;               // (t*J + j)*I + i
    float v = 0.0f;
    if (k < KGRID) v = fc[rest * KGRID + k];
    w[idx] = (_Float16)v;
}

// ---- main: one wave = 16 batch rows x all 64 outputs ----
__global__ void __launch_bounds__(256) fkan_main(
    const float* __restrict__ x, const float* __restrict__ bias,
    const _Float16* __restrict__ w, float* __restrict__ out, int nrows)
{
    const int lane  = threadIdx.x & 31;
    const int wid   = threadIdx.x >> 5;
    const int tile  = blockIdx.x * (blockDim.x >> 5) + wid;
    const int bbase = tile * 16;
    if (bbase >= nrows) return;          // wave-uniform: EXEC stays all-ones for WMMA

    const int  l15   = lane & 15;
    const bool hi    = lane >= 16;
    const int  base8 = hi ? 8 : 0;       // A-layout: lanes>=16 own K-halfs 8..15 / 24..31
    const int  row   = bbase + l15;
    const int  jstr  = IN_DIM * KPAD;    // halves per j row in prepped coeff array

    v8f acc0 = {}, acc1 = {}, acc2 = {}, acc3 = {};

    for (int i = 0; i < IN_DIM; ++i) {
        const float xv = x[row * IN_DIM + i];

        // init 8 (cos,sin) pairs for multipliers m = base8 + h + 1  (k is 1-based)
        float c[8], s[8];
        #pragma unroll
        for (int h = 0; h < 8; ++h) {
            float ph = xv * (float)(base8 + h + 1);
            __sincosf(ph, &s[h], &c[h]);
        }
        // step rotation: advance all harmonics by +16 via angle addition
        float S16, C16;
        __sincosf(16.0f * xv, &S16, &C16);

        // B source: halfs 0..7 -> cos coeffs (t=0) k=base8..base8+7, halfs 8..15 -> sin coeffs (t=1) same k,
        // row j = jt*16 + l15
        const _Float16* wl = w + ((l15) * IN_DIM + i) * KPAD + base8;

        #pragma unroll
        for (int step = 0; step < STEPS; ++step) {
            // A fragment 16x32 f16: halfs 0..7 = cos, halfs 8..15 = sin (this lane's 8 harmonics)
            v16h a;
            #pragma unroll
            for (int h = 0; h < 8; ++h) {
                a[h]     = (_Float16)c[h];
                a[8 + h] = (_Float16)s[h];
            }
            const _Float16* ws = wl + step * 16;
            const int so = OUT_DIM * IN_DIM * KPAD;
            v16h b0, b1, b2, b3;
            #pragma unroll
            for (int e = 0; e < 8; ++e) {
                b0[e] = ws[e];             b0[8 + e] = ws[so + e];
                b1[e] = ws[16 * jstr + e]; b1[8 + e] = ws[so + 16 * jstr + e];
                b2[e] = ws[32 * jstr + e]; b2[8 + e] = ws[so + 32 * jstr + e];
                b3[e] = ws[48 * jstr + e]; b3[8 + e] = ws[so + 48 * jstr + e];
            }

            acc0 = __builtin_amdgcn_wmma_f32_16x16x32_f16(false, a, false, b0, (short)0, acc0, false, false);
            acc1 = __builtin_amdgcn_wmma_f32_16x16x32_f16(false, a, false, b1, (short)0, acc1, false, false);
            acc2 = __builtin_amdgcn_wmma_f32_16x16x32_f16(false, a, false, b2, (short)0, acc2, false, false);
            acc3 = __builtin_amdgcn_wmma_f32_16x16x32_f16(false, a, false, b3, (short)0, acc3, false, false);

            if (step != STEPS - 1) {
                #pragma unroll
                for (int h = 0; h < 8; ++h) {
                    float cn = __builtin_fmaf(c[h], C16, -(s[h] * S16));
                    float sn = __builtin_fmaf(s[h], C16,  (c[h] * S16));
                    c[h] = cn; s[h] = sn;
                }
            }
        }
    }

    // C/D layout: lane l15 = column n; VGPR v = row M = v + (hi ? 8 : 0)
    const int r0 = bbase + (hi ? 8 : 0);
    const float bj0 = bias[l15], bj1 = bias[16 + l15], bj2 = bias[32 + l15], bj3 = bias[48 + l15];
    #pragma unroll
    for (int v = 0; v < 8; ++v) {
        float* o = out + (size_t)(r0 + v) * OUT_DIM;
        o[l15]      = acc0[v] + bj0;
        o[16 + l15] = acc1[v] + bj1;
        o[32 + l15] = acc2[v] + bj2;
        o[48 + l15] = acc3[v] + bj3;
    }
}

extern "C" void kernel_launch(void* const* d_in, const int* in_sizes, int n_in,
                              void* d_out, int out_size, void* d_ws, size_t ws_size,
                              hipStream_t stream) {
    const float* x    = (const float*)d_in[0];   // [N, 64]
    const float* fc   = (const float*)d_in[1];   // [2, 64, 64, 100]
    const float* bias = (const float*)d_in[2];   // [64]
    float* out        = (float*)d_out;           // [N, 64]
    _Float16* w       = (_Float16*)d_ws;         // [2][64][64][112] f16 = 1,835,008 B

    const int nrows = in_sizes[0] / IN_DIM;      // 20000

    const int total = 2 * OUT_DIM * IN_DIM * KPAD;
    fkan_prep<<<(total + 255) / 256, 256, 0, stream>>>(fc, w, total);

    const int tiles = (nrows + 15) / 16;         // 1250
    const int wpb   = 256 / 32;                  // 8 waves per block
    fkan_main<<<(tiles + wpb - 1) / wpb, 256, 0, stream>>>(x, bias, w, out, nrows);
}
